// AttentionVAE_87153476370724
// MI455X (gfx1250) — matrix operand order repaired and checked by reading, not compiled, not run on hardware
//
#include <hip/hip_runtime.h>

typedef __attribute__((ext_vector_type(16))) _Float16 v16h;
typedef __attribute__((ext_vector_type(8)))  float    v8f;
typedef _Float16 h16;

#define BTOT 262144
#define DIN  196

// ---- pre-swizzled weight fragment table (units of 512 halves = one 32x16 B tile) ----
#define FR_FA1 0    // 7nt x 7kt = 49
#define FR_FA2 49   // 13 x 4 = 52
#define FR_E1  101  // 6 x 7  = 42
#define FR_E2  143  // 4 x 3  = 12
#define FR_E3  155  // 2 x 2  = 4
#define FR_EMV 159  // 1      (e_wm|e_wv packed N=16)
#define FR_D1  160  // 2 x 1
#define FR_D2  162  // 4 x 1
#define FR_D3  166  // 6 x 2
#define FR_D4  178  // 13 x 3
#define FR_TOTAL 217

// ---- per-wave LDS layout (bytes) ----
#define XOFF 0        // x / x_attn : 16x224 f16 = 7168B
#define TOFF 7168     // fa tanh (16x128) then decoder ping-pong region, 6144B
#define AOFF 13312    // a (16x224) then h1|h2|h3|z, 7168B
#define FOFF 20480    // f32 scratch: losses[16*3]
#define WAVE_BYTES 20736
#define NWAVE 8
#define BLK_ROWS (NWAVE * 16)

// ---------------- fast native transcendentals ----------------
__device__ __forceinline__ float frcp(float x) { return __builtin_amdgcn_rcpf(x); }
__device__ __forceinline__ float fsigmoid(float x) {
  // exp -> inf => rcp -> 0 ; exp -> 0 => 1.  No branches needed.
  return frcp(1.0f + __expf(-x));
}
__device__ __forceinline__ float ftanh(float x) {
  float xc = fminf(fmaxf(x, -15.0f), 15.0f);   // tanh saturates; avoid inf/inf
  float e  = __expf(2.0f * xc);
  return (e - 1.0f) * frcp(e + 1.0f);
}

// ---------------- fragment loaders ----------------
__device__ __forceinline__ v16h load_afrag(const h16* buf, int stride, int kt, int lane) {
  // 16-bit A 16x32 layout: lane<16 -> K 0..7 & 16..23 ; lane>=16 -> K 8..15 & 24..31
  const h16* p = buf + (lane & 15) * stride + kt * 32 + ((lane & 16) ? 8 : 0);
  union { v16h v; float4 q[2]; } u;
  u.q[0] = *(const float4*)(p);
  u.q[1] = *(const float4*)(p + 16);
  return u.v;
}

__device__ __forceinline__ v16h load_bfrag(const h16* wf, int frag, int lane) {
  const h16* p = wf + ((size_t)frag << 9) + (lane << 4);
  union { v16h v; float4 q[2]; } u;
  u.q[0] = *(const float4*)(p);
  u.q[1] = *(const float4*)(p + 8);
  return u.v;
}

template<int NKT>
__device__ __forceinline__ v8f gemm16(const h16* abuf, int astride, const h16* wf,
                                      int nt, int lane) {
  v8f c = {};
#pragma unroll
  for (int kt = 0; kt < NKT; ++kt) {
    v16h a = load_afrag(abuf, astride, kt, lane);
    v16h b = load_bfrag(wf, nt * NKT + kt, lane);
    c = __builtin_amdgcn_wmma_f32_16x16x32_f16(false, a, false, b, (short)0, c,
                                               false, false);
  }
  return c;
}

// GEMM + bias + activation -> f16 LDS buffer.  ACT: 0=leaky_relu(0.1), 1=tanh
template<int NKT, int ACT>
__device__ __forceinline__ void layer(const h16* abuf, int astride, const h16* wf,
                                      const float* bias, int Nreal, int nnt,
                                      h16* obuf, int ostride, int lane) {
  const int j = lane & 15, rb = (lane & 16) ? 8 : 0;
  for (int nt = 0; nt < nnt; ++nt) {
    v8f c = gemm16<NKT>(abuf, astride, wf, nt, lane);
    const int col = nt * 16 + j;
    const float bv = (col < Nreal) ? bias[col] : 0.0f;
#pragma unroll
    for (int r = 0; r < 8; ++r) {
      float v = c[r] + bv;
      v = (ACT == 1) ? ftanh(v) : (v > 0.0f ? v : 0.1f * v);
      obuf[(rb + r) * ostride + col] = (h16)v;   // padded cols store act(0)=0
    }
  }
}

// ---------------- misc device helpers ----------------
__device__ __forceinline__ unsigned hashu(unsigned x) {
  x ^= x >> 16; x *= 0x7feb352dU; x ^= x >> 15; x *= 0x846ca68bU; x ^= x >> 16;
  return x;
}
__device__ __forceinline__ float gauss_rng(unsigned row, unsigned jj, unsigned i) {
  unsigned s = row * 0x9E3779B9u + jj * 0x85EBCA6Bu + i * 0xC2B2AE35u + 0x2545F491u;
  unsigned u1 = hashu(s), u2 = hashu(s ^ 0x68E31DA4u);
  float f1 = ((float)(u1 >> 8) + 1.0f) * (1.0f / 16777217.0f);
  float f2 = (float)(u2 >> 8) * (1.0f / 16777216.0f);
  return sqrtf(-2.0f * __logf(f1)) * __cosf(6.28318530717958648f * f2);
}

struct MainParams {
  const float *x, *fa_b1, *fa_b2, *e_b1, *e_b2, *e_b3, *e_bm, *e_bv;
  const float *d_b1, *d_b2, *d_b3, *d_b4, *la_w, *la_b;
  const h16* wpre;
  float *outRecon, *outMean, *outLogvar, *outAttn, *partials;
};

// ---------------- main fused kernel: one wave = 16 rows ----------------
__global__ __launch_bounds__(256) void vae_main(MainParams p) {
  extern __shared__ char smem[];
  const int lane = threadIdx.x & 31;
  const int wv   = threadIdx.x >> 5;
  const int j    = lane & 15;
  const int rb   = (lane & 16) ? 8 : 0;
  const int row0 = blockIdx.x * BLK_ROWS + wv * 16;

  char* wbase = smem + wv * WAVE_BYTES;
  h16* X  = (h16*)(wbase + XOFF);     // 16 x 224
  h16* T  = (h16*)(wbase + TOFF);     // 16 x 128 (fa) / decoder region
  h16* A  = (h16*)(wbase + AOFF);     // 16 x 224 (a) -> later h1/h2/h3/z
  h16* H1 = A;                        // 16 x 96
  h16* H2 = A + 1536;                 // 16 x 64
  h16* H3 = A + 2560;                 // 16 x 32
  h16* Z  = A + 3072;                 // 16 x 32
  h16* D1 = T;                        // 16 x 32
  h16* D2 = T + 512;                  // 16 x 64
  h16* D3 = T + 1536;                 // 16 x 96
  float* LL    = (float*)(wbase + FOFF);              // losses[16][3]
  float* wpart = (float*)(smem + NWAVE * WAVE_BYTES); // [NWAVE]

  // zero the whole per-wave region (gives us the K-padding zero columns)
  {
    float4 z4 = make_float4(0.f, 0.f, 0.f, 0.f);
    float4* q = (float4*)wbase;
    for (int t = lane; t < WAVE_BYTES / 16; t += 32) q[t] = z4;
  }

  // load x rows -> f16 LDS (cols 196..223 stay zero)
  for (int r = 0; r < 16; ++r) {
    const float* xr = p.x + (size_t)(row0 + r) * DIN;
    for (int c = lane; c < DIN; c += 32) X[r * 224 + c] = (h16)xr[c];
  }

  // ---- feature attention ----
  layer<7, 1>(X, 224, p.wpre + ((size_t)FR_FA1 << 9), p.fa_b1, 98, 7, T, 128, lane);

  float rs[8];
#pragma unroll
  for (int r = 0; r < 8; ++r) rs[r] = 0.f;
  for (int nt = 0; nt < 13; ++nt) {
    v8f c = gemm16<4>(T, 128, p.wpre + ((size_t)FR_FA2 << 9), nt, lane);
    const int col = nt * 16 + j;
    const bool ok = col < DIN;
    const float bv = ok ? p.fa_b2[col] : 0.f;
#pragma unroll
    for (int r = 0; r < 8; ++r) {
      float av = fsigmoid(c[r] + bv);
      if (ok) { A[(rb + r) * 224 + col] = (h16)av; rs[r] += av; }
    }
  }
#pragma unroll
  for (int m = 1; m <= 8; m <<= 1)
#pragma unroll
    for (int r = 0; r < 8; ++r) rs[r] += __shfl_xor(rs[r], m, 32);
  float im[8];
#pragma unroll
  for (int r = 0; r < 8; ++r) im[r] = frcp(rs[r] * (1.f / 196.f) + 1e-8f);

  for (int nt = 0; nt < 13; ++nt) {
    const int col = nt * 16 + j;
    if (col < DIN) {
#pragma unroll
      for (int r = 0; r < 8; ++r) {
        const int row = rb + r;
        float at = (float)A[row * 224 + col] * im[r];
        p.outAttn[(size_t)(row0 + row) * DIN + col] = at;
        X[row * 224 + col] = (h16)((float)X[row * 224 + col] * at);  // x_attn in place
      }
    }
  }

  // ---- encoder ----
  layer<7, 0>(X, 224, p.wpre + ((size_t)FR_E1 << 9), p.e_b1, 96, 6, H1, 96, lane);
  layer<3, 0>(H1, 96, p.wpre + ((size_t)FR_E2 << 9), p.e_b2, 64, 4, H2, 64, lane);
  layer<2, 0>(H2, 64, p.wpre + ((size_t)FR_E3 << 9), p.e_b3, 32, 2, H3, 32, lane);

  // mean | logvar packed in one N=16 tile (cols 0..7 mean, 8..15 logvar)
  float mu[8], sg[8];
  {
    v8f c = gemm16<1>(H3, 32, p.wpre + ((size_t)FR_EMV << 9), 0, lane);
    const float bv = (j < 8) ? p.e_bm[j] : p.e_bv[j - 8];
#pragma unroll
    for (int r = 0; r < 8; ++r) {
      float v = c[r] + bv;
      if (j < 8) p.outMean[(size_t)(row0 + rb + r) * 8 + j] = v;
      else       p.outLogvar[(size_t)(row0 + rb + r) * 8 + (j - 8)] = v;
      float lv = __shfl(v, lane + 8, 32);   // grab logvar from partner lane (valid for j<8)
      mu[r] = v;
      sg[r] = __expf(0.5f * lv);
    }
  }

  // re-zero Z slot (it was clobbered by the a-buffer); cols 8..31 must be 0
  {
    float4 z4 = make_float4(0.f, 0.f, 0.f, 0.f);
    float4* q = (float4*)Z;
    for (int t = lane; t < 64; t += 32) q[t] = z4;
  }

  // ---- decoder x4 (run 0 -> recon output; runs 1..3 -> layer losses) ----
  for (int i = 0; i < 4; ++i) {
    if (j < 8) {
#pragma unroll
      for (int r = 0; r < 8; ++r) {
        const int row = rb + r;
        float eps = gauss_rng((unsigned)(row0 + row), (unsigned)j, (unsigned)i);
        Z[row * 32 + j] = (h16)(mu[r] + eps * sg[r]);
      }
    }
    layer<1, 0>(Z,  32, p.wpre + ((size_t)FR_D1 << 9), p.d_b1, 32, 2, D1, 32, lane);
    layer<1, 0>(D1, 32, p.wpre + ((size_t)FR_D2 << 9), p.d_b2, 64, 4, D2, 64, lane);
    layer<2, 0>(D2, 64, p.wpre + ((size_t)FR_D3 << 9), p.d_b3, 96, 6, D3, 96, lane);

    const int nnt = (i == 0) ? 13 : (i == 1 ? 6 : (i == 2 ? 4 : 2));
    const int dim = (i == 1) ? 96 : (i == 2 ? 64 : 32);
    const h16* H  = (i == 1) ? H1 : (i == 2 ? H2 : H3);

    float ls[8];
#pragma unroll
    for (int r = 0; r < 8; ++r) ls[r] = 0.f;

    for (int nt = 0; nt < nnt; ++nt) {
      v8f c = gemm16<3>(D3, 96, p.wpre + ((size_t)FR_D4 << 9), nt, lane);
      const int col = nt * 16 + j;
      const float bv = (col < DIN) ? p.d_b4[col] : 0.f;
#pragma unroll
      for (int r = 0; r < 8; ++r) {
        float s = fsigmoid(c[r] + bv);
        if (i == 0) {
          if (col < DIN) p.outRecon[(size_t)(row0 + rb + r) * DIN + col] = s;
        } else if (col < dim) {
          float d = (float)H[(rb + r) * dim + col] - s;
          ls[r] += d * d;
        }
      }
    }
    if (i > 0) {
#pragma unroll
      for (int m = 1; m <= 8; m <<= 1)
#pragma unroll
        for (int r = 0; r < 8; ++r) ls[r] += __shfl_xor(ls[r], m, 32);
      if (j == 0) {
#pragma unroll
        for (int r = 0; r < 8; ++r) LL[(rb + r) * 3 + (i - 1)] = ls[r] / (float)dim;
      }
    }
  }

  // ---- per-row layer attention + weighted loss ----
  float wsum = 0.f;
  if (lane < 16) {
    float l0 = LL[lane * 3 + 0], l1 = LL[lane * 3 + 1], l2 = LL[lane * 3 + 2];
    float lg[3];
#pragma unroll
    for (int t = 0; t < 3; ++t)
      lg[t] = l0 * p.la_w[0 * 3 + t] + l1 * p.la_w[1 * 3 + t] + l2 * p.la_w[2 * 3 + t] +
              p.la_b[t];
    float mx = fmaxf(lg[0], fmaxf(lg[1], lg[2]));
    float e0 = __expf(lg[0] - mx), e1 = __expf(lg[1] - mx), e2 = __expf(lg[2] - mx);
    wsum = (l0 * e0 + l1 * e1 + l2 * e2) * frcp(e0 + e1 + e2);
  }
#pragma unroll
  for (int m = 1; m <= 8; m <<= 1) wsum += __shfl_xor(wsum, m, 32);
  if (lane == 0) wpart[wv] = wsum;
  __syncthreads();
  if (threadIdx.x == 0) {
    float s = 0.f;
#pragma unroll
    for (int w = 0; w < NWAVE; ++w) s += wpart[w];
    p.partials[blockIdx.x] = s;
  }
}

// ---------------- weight pre-swizzle into WMMA B-fragment layout ----------------
__global__ void prep_w(const float* __restrict__ W, int K, int N, int nkt,
                       h16* __restrict__ out) {
  const int lane = threadIdx.x;               // 32 threads
  const int nt = blockIdx.x / nkt, kt = blockIdx.x % nkt;
  const int n  = nt * 16 + (lane & 15);
  const int kb = kt * 32 + ((lane & 16) ? 8 : 0);
  h16* o = out + (size_t)blockIdx.x * 512 + lane * 16;
#pragma unroll
  for (int t = 0; t < 16; ++t) {
    const int k = kb + (t < 8 ? t : t + 8);
    float v = (k < K && n < N) ? W[(size_t)k * N + n] : 0.f;
    o[t] = (h16)v;
  }
}

__global__ void prep_mv(const float* __restrict__ Wm, const float* __restrict__ Wv,
                        h16* __restrict__ out) {
  const int lane = threadIdx.x;
  const int n  = lane & 15;
  const int kb = (lane & 16) ? 8 : 0;
  h16* o = out + lane * 16;
#pragma unroll
  for (int t = 0; t < 16; ++t) {
    const int k = kb + (t < 8 ? t : t + 8);
    float v = (n < 8) ? Wm[k * 8 + n] : Wv[k * 8 + (n - 8)];
    o[t] = (h16)v;
  }
}

__global__ void finalize_k(const float* __restrict__ partials, int n, float* out) {
  __shared__ float sh[256];
  float s = 0.f;
  for (int i = threadIdx.x; i < n; i += 256) s += partials[i];
  sh[threadIdx.x] = s;
  __syncthreads();
  for (int st = 128; st > 0; st >>= 1) {
    if (threadIdx.x < st) sh[threadIdx.x] += sh[threadIdx.x + st];
    __syncthreads();
  }
  if (threadIdx.x == 0) out[0] = sh[0] / (float)BTOT;
}

// ---------------- host ----------------
extern "C" void kernel_launch(void* const* d_in, const int* in_sizes, int n_in,
                              void* d_out, int out_size, void* d_ws, size_t ws_size,
                              hipStream_t stream) {
  (void)in_sizes; (void)n_in; (void)out_size; (void)ws_size;

  const float* x     = (const float*)d_in[0];
  const float* fa_w1 = (const float*)d_in[1];
  const float* fa_b1 = (const float*)d_in[2];
  const float* fa_w2 = (const float*)d_in[3];
  const float* fa_b2 = (const float*)d_in[4];
  const float* e_w1  = (const float*)d_in[5];
  const float* e_b1  = (const float*)d_in[6];
  const float* e_w2  = (const float*)d_in[7];
  const float* e_b2  = (const float*)d_in[8];
  const float* e_w3  = (const float*)d_in[9];
  const float* e_b3  = (const float*)d_in[10];
  const float* e_wm  = (const float*)d_in[11];
  const float* e_bm  = (const float*)d_in[12];
  const float* e_wv  = (const float*)d_in[13];
  const float* e_bv  = (const float*)d_in[14];
  const float* d_w1  = (const float*)d_in[15];
  const float* d_b1  = (const float*)d_in[16];
  const float* d_w2  = (const float*)d_in[17];
  const float* d_b2  = (const float*)d_in[18];
  const float* d_w3  = (const float*)d_in[19];
  const float* d_b3  = (const float*)d_in[20];
  const float* d_w4  = (const float*)d_in[21];
  const float* d_b4  = (const float*)d_in[22];
  const float* la_w  = (const float*)d_in[23];
  const float* la_b  = (const float*)d_in[24];

  h16*   wpre     = (h16*)d_ws;
  float* partials = (float*)((char*)d_ws + 240 * 1024);   // past 217*1024B of fragments

  prep_w<<<49, 32, 0, stream>>>(fa_w1, 196, 98, 7, wpre + ((size_t)FR_FA1 << 9));
  prep_w<<<52, 32, 0, stream>>>(fa_w2, 98, 196, 4, wpre + ((size_t)FR_FA2 << 9));
  prep_w<<<42, 32, 0, stream>>>(e_w1, 196, 96, 7, wpre + ((size_t)FR_E1 << 9));
  prep_w<<<12, 32, 0, stream>>>(e_w2, 96, 64, 3, wpre + ((size_t)FR_E2 << 9));
  prep_w<<< 4, 32, 0, stream>>>(e_w3, 64, 32, 2, wpre + ((size_t)FR_E3 << 9));
  prep_mv<<<1, 32, 0, stream>>>(e_wm, e_wv, wpre + ((size_t)FR_EMV << 9));
  prep_w<<< 2, 32, 0, stream>>>(d_w1, 8, 32, 1, wpre + ((size_t)FR_D1 << 9));
  prep_w<<< 4, 32, 0, stream>>>(d_w2, 32, 64, 1, wpre + ((size_t)FR_D2 << 9));
  prep_w<<<12, 32, 0, stream>>>(d_w3, 64, 96, 2, wpre + ((size_t)FR_D3 << 9));
  prep_w<<<39, 32, 0, stream>>>(d_w4, 96, 196, 3, wpre + ((size_t)FR_D4 << 9));

  float* out = (float*)d_out;
  MainParams prm;
  prm.x = x; prm.fa_b1 = fa_b1; prm.fa_b2 = fa_b2;
  prm.e_b1 = e_b1; prm.e_b2 = e_b2; prm.e_b3 = e_b3; prm.e_bm = e_bm; prm.e_bv = e_bv;
  prm.d_b1 = d_b1; prm.d_b2 = d_b2; prm.d_b3 = d_b3; prm.d_b4 = d_b4;
  prm.la_w = la_w; prm.la_b = la_b;
  prm.wpre = wpre;
  prm.outRecon  = out;
  prm.outMean   = out + (size_t)BTOT * DIN;
  prm.outLogvar = prm.outMean + (size_t)BTOT * 8;
  prm.outAttn   = prm.outLogvar + (size_t)BTOT * 8;
  prm.partials  = partials;

  const int nBlocks = BTOT / BLK_ROWS;                    // 2048
  const size_t smemBytes = (size_t)NWAVE * WAVE_BYTES + 64;
  hipFuncSetAttribute((const void*)vae_main,
                      hipFuncAttributeMaxDynamicSharedMemorySize, (int)smemBytes);
  vae_main<<<nBlocks, 256, smemBytes, stream>>>(prm);

  float* outScalar = prm.outAttn + (size_t)BTOT * DIN;
  finalize_k<<<1, 256, 0, stream>>>(partials, nBlocks, outScalar);
}
